// VAE_GAT_72335839200002
// MI455X (gfx1250) — compile-verified, matrix operand order checked by reading, not run on hardware
//
#include <hip/hip_runtime.h>

typedef __attribute__((ext_vector_type(16))) _Float16 v16h;
typedef __attribute__((ext_vector_type(8)))  float    v8f;

#define NN 10000
#define EE 320000
#define ET 330000   // E + N self loops
#define HH 4
#define TM 625      // NN/16 row tiles

// ---------- helpers ----------
__device__ __forceinline__ unsigned encf(float f) {
    unsigned u = __float_as_uint(f);
    return (u & 0x80000000u) ? ~u : (u | 0x80000000u);
}
__device__ __forceinline__ float decf(unsigned u) {
    unsigned v = (u & 0x80000000u) ? (u & 0x7FFFFFFFu) : ~u;
    return __uint_as_float(v);
}

// CDNA5 async DMA: 16B global -> LDS, tracked by ASYNCcnt (ISA 08, opcode 98).
// lds_off = wave-relative LDS byte address (low 32 bits of generic shared ptr).
__device__ __forceinline__ void async_b128(unsigned lds_off, unsigned long long gaddr) {
    asm volatile("global_load_async_to_lds_b128 %0, %1, off"
                 :: "v"(lds_off), "v"(gaddr) : "memory");
}
__device__ __forceinline__ void wait_async1() {
    asm volatile("s_wait_asynccnt 0x1" ::: "memory");
}
__device__ __forceinline__ void wait_async0() {
    asm volatile("s_wait_asynccnt 0x0" ::: "memory");
}

__global__ void fill_f32_kernel(float* p, float v, int n) {
    int i = blockIdx.x * blockDim.x + threadIdx.x;
    if (i < n) p[i] = v;
}
__global__ void fill_u32_kernel(unsigned* p, unsigned v, int n) {
    int i = blockIdx.x * blockDim.x + threadIdx.x;
    if (i < n) p[i] = v;
}
__global__ void tohalf_kernel(const float* __restrict__ s, _Float16* __restrict__ d, int n) {
    int i = blockIdx.x * blockDim.x + threadIdx.x;
    if (i < n) d[i] = (_Float16)s[i];
}
// BT[n][k] = (f16) B[k][n]   (pre-transpose weights so async chunks are contiguous)
__global__ void convT_kernel(const float* __restrict__ B, _Float16* __restrict__ BT,
                             int K, int Nc) {
    int i = blockIdx.x * blockDim.x + threadIdx.x;
    if (i >= K * Nc) return;
    int n = i / K, k = i - (i / K) * K;
    BT[i] = (_Float16)B[(size_t)k * Nc + n];
}

// ============================================================================
// WMMA GEMM: C[M,Nc] = A[M,K] @ B[K,Nc] (+bias)(+relu); B given pre-transposed
// f16 as BT[Nc][K]. Block = 8 waves -> 128 rows x 64 cols. B chunks are
// double-buffered into LDS via global_load_async_to_lds_b128 (1 op/thread/chunk),
// overlapping DMA with WMMA. A fragment reused over 4 j-tiles.
// ============================================================================
__global__ void wmma_gemm_kernel(const float* __restrict__ A, const _Float16* __restrict__ BT,
                                 const float* __restrict__ bias, float* __restrict__ C,
                                 int M, int K, int Nc, int relu)
{
    __shared__ _Float16 LB[2][64][72];       // [buf][col][k], 144B row stride (aligned, bank-spread)
    const int tid  = threadIdx.x;
    const int lane = tid & 31;
    const int w    = tid >> 5;               // wave in block, 0..7
    const int tMt  = M >> 4;
    const int jg   = blockIdx.x;             // 64-col group
    const int ig   = blockIdx.y;             // 128-row group (8 i-tiles)
    const int ti   = ig * 8 + w;
    const bool valid = (ti < tMt);           // wave-uniform
    const int jbase = jg * 64;
    const int half = lane >> 4;
    const int l15  = lane & 15;
    const int arow = ti * 16 + l15;

    // staging map: thread -> (col, 16B quarter of a 64B k-chunk row)
    const int scol = tid >> 2;               // 0..63
    const int sq   = tid & 3;                // 0..3
    const unsigned long long gsrc =
        (unsigned long long)(size_t)(BT + (size_t)(jbase + scol) * K + sq * 8);
    const unsigned lds0 = (unsigned)(size_t)&LB[0][scol][sq * 8];
    const unsigned lds1 = (unsigned)(size_t)&LB[1][scol][sq * 8];

    const int nch = K >> 5;
    async_b128(lds0, gsrc);                  // prologue: chunk 0

    v8f acc[4] = {};
    for (int ch = 0; ch < nch; ++ch) {
        if (ch + 1 < nch) {                  // issue next chunk (other buffer)
            async_b128(((ch + 1) & 1) ? lds1 : lds0,
                       gsrc + (unsigned long long)(ch + 1) * 64u);
            wait_async1();                   // chunk ch landed (1 newer op outstanding)
        } else {
            wait_async0();
        }
        __syncthreads();                     // all waves' portions visible
        if (valid) {
            v16h a;
            const float* Ap = A + (size_t)arow * K + ch * 32 + half * 8;
#pragma unroll
            for (int i = 0; i < 8; ++i) {    // ISA 7.12.2 16-bit A 16x32 layout
                a[i]     = (_Float16)Ap[i];
                a[i + 8] = (_Float16)Ap[i + 16];
            }
#pragma unroll
            for (int jt = 0; jt < 4; ++jt) {
                v16h b;
                const _Float16* Bp = &LB[ch & 1][jt * 16 + l15][half * 8];
#pragma unroll
                for (int i = 0; i < 8; ++i) { b[i] = Bp[i]; b[i + 8] = Bp[i + 16]; }
                acc[jt] = __builtin_amdgcn_wmma_f32_16x16x32_f16(false, a, false, b,
                                                                 (short)0, acc[jt],
                                                                 false, false);
            }
        }
        __syncthreads();                     // done reading before buffer is re-filled
    }
    if (valid) {
        const int rbase = ti * 16 + half * 8;
#pragma unroll
        for (int jt = 0; jt < 4; ++jt) {
            const int col  = jbase + jt * 16 + l15;
            const float bv = bias ? bias[col] : 0.0f;
#pragma unroll
            for (int r = 0; r < 8; ++r) {
                float v = acc[jt][r] + bv;
                if (relu) v = fmaxf(v, 0.0f);
                C[(size_t)(rbase + r) * Nc + col] = v;
            }
        }
    }
}

// ---------- attention scores: al/ar[n*H+h] = <feat[n,h,:], a_{src,dst}[h,:]> ----------
__global__ void att_scores_kernel(const float* __restrict__ feat, const float* __restrict__ as_,
                                  const float* __restrict__ ad_, float* __restrict__ al,
                                  float* __restrict__ ar, int NHtot, int C)
{
    const int lane = threadIdx.x & 31;
    const int wave = blockIdx.x * (blockDim.x >> 5) + (threadIdx.x >> 5);
    if (wave >= NHtot) return;
    const int h = wave % HH;
    const float* f = feat + (size_t)wave * C;    // feat is [N, H*C]; wave = n*H+h
    const float* s = as_ + h * C;
    const float* d = ad_ + h * C;
    float sl = 0.f, sr = 0.f;
    for (int c = lane; c < C; c += 32) { float fv = f[c]; sl += fv * s[c]; sr += fv * d[c]; }
    for (int off = 16; off; off >>= 1) { sl += __shfl_xor(sl, off, 32); sr += __shfl_xor(sr, off, 32); }
    if (lane == 0) { al[wave] = sl; ar[wave] = sr; }
}

__device__ __forceinline__ void edge_src_dst(const int* ei, int e, int& src, int& dst) {
    if (e < EE) { src = ei[e]; dst = ei[EE + e]; }
    else        { src = dst = e - EE; }          // self loops appended
}

// ---------- segment max of leaky_relu(al[src]+ar[dst]) over dst ----------
__global__ void edge_max_kernel(const int* __restrict__ ei, const float* __restrict__ al,
                                const float* __restrict__ ar, unsigned* __restrict__ mx)
{
    int idx = blockIdx.x * blockDim.x + threadIdx.x;
    if (idx >= ET * HH) return;
    int e = idx / HH, h = idx % HH, src, dst;
    edge_src_dst(ei, e, src, dst);
    float v = al[src * HH + h] + ar[dst * HH + h];
    v = v > 0.f ? v : 0.2f * v;
    atomicMax(mx + dst * HH + h, encf(v));
}

// ---------- ex = exp(e - m[dst]); denom[dst] += ex ----------
__global__ void edge_exp_kernel(const int* __restrict__ ei, const float* __restrict__ al,
                                const float* __restrict__ ar, const unsigned* __restrict__ mx,
                                float* __restrict__ exbuf, float* __restrict__ denom)
{
    int idx = blockIdx.x * blockDim.x + threadIdx.x;
    if (idx >= ET * HH) return;
    int e = idx / HH, h = idx % HH, src, dst;
    edge_src_dst(ei, e, src, dst);
    float v = al[src * HH + h] + ar[dst * HH + h];
    v = v > 0.f ? v : 0.2f * v;
    float ex = __expf(v - decf(mx[dst * HH + h]));
    exbuf[idx] = ex;
    atomicAdd(denom + dst * HH + h, ex);
}

// ---------- acc[dst,h,:] += alpha * feat[src,h,:] ; wave per (edge,head), b128 loads ----------
__global__ void edge_scatter_kernel(const int* __restrict__ ei, const float* __restrict__ exbuf,
                                    const float* __restrict__ denom, const float* __restrict__ feat,
                                    float* __restrict__ acc, int C)
{
    const int lane = threadIdx.x & 31;
    const int wave = blockIdx.x * (blockDim.x >> 5) + (threadIdx.x >> 5);
    if (wave >= ET * HH) return;
    int e = wave / HH, h = wave % HH, src, dst;
    edge_src_dst(ei, e, src, dst);
    const float alpha = exbuf[wave] / denom[dst * HH + h];
    const float4* f = (const float4*)(feat + ((size_t)src * HH + h) * C);
    float* o        = acc + ((size_t)dst * HH + h) * C;
    const int C4 = C >> 2;
    for (int c4 = lane; c4 < C4; c4 += 32) {
        float4 v = f[c4];
        const int c = c4 << 2;
        atomicAdd(o + c + 0, alpha * v.x);
        atomicAdd(o + c + 1, alpha * v.y);
        atomicAdd(o + c + 2, alpha * v.z);
        atomicAdd(o + c + 3, alpha * v.w);
    }
}

// ---------- out[n,c] = (relu)( mean_h acc[n,h,c] + bias[c] ) ----------
__global__ void head_mean_kernel(const float* __restrict__ acc, const float* __restrict__ bias,
                                 float* __restrict__ out, int C, int relu)
{
    int idx = blockIdx.x * blockDim.x + threadIdx.x;
    if (idx >= NN * C) return;
    int n = idx / C, c = idx % C;
    float s = 0.f;
#pragma unroll
    for (int h = 0; h < HH; ++h) s += acc[((size_t)n * HH + h) * C + c];
    float v = s * (1.0f / HH) + bias[c];
    if (relu) v = fmaxf(v, 0.f);
    out[idx] = v;
}

// ---------- z = mu + eps * exp(0.5*logvar); hash-based Box-Muller eps ----------
__global__ void reparam_kernel(const float* __restrict__ mu, const float* __restrict__ lv,
                               float* __restrict__ z, int n)
{
    int i = blockIdx.x * blockDim.x + threadIdx.x;
    if (i >= n) return;
    unsigned s1 = (unsigned)i * 747796405u + 2891336453u;
    unsigned w1 = ((s1 >> ((s1 >> 28) + 4u)) ^ s1) * 277803737u; w1 = (w1 >> 22) ^ w1;
    unsigned s2 = ((unsigned)i ^ 0x9e3779b9u) * 747796405u + 2891336453u;
    unsigned w2 = ((s2 >> ((s2 >> 28) + 4u)) ^ s2) * 277803737u; w2 = (w2 >> 22) ^ w2;
    float u1 = (float)(w1 | 1u) * 2.3283064365386963e-10f;
    float u2 = (float)w2 * 2.3283064365386963e-10f;
    float eps = sqrtf(fmaxf(-2.f * __logf(u1), 0.f)) * __cosf(6.283185307f * u2);
    z[i] = mu[i] + eps * __expf(0.5f * lv[i]);
}

// ---------- sq[n] = ||X[n,:]||^2 (f32 source for accuracy) ----------
__global__ void sqnorm_kernel(const float* __restrict__ X, float* __restrict__ sq, int F)
{
    const int lane = threadIdx.x & 31;
    const int wave = blockIdx.x * (blockDim.x >> 5) + (threadIdx.x >> 5);
    if (wave >= NN) return;
    const float* x = X + (size_t)wave * F;
    float s = 0.f;
    for (int c = lane; c < F; c += 32) s += x[c] * x[c];
    for (int off = 16; off; off >>= 1) s += __shfl_xor(s, off, 32);
    if (lane == 0) sq[wave] = s;
}

// ============================================================================
// pd[i,j] = sqrt(max(sq_i+sq_j-2*<x_i,x_j>, 1e-12)), Xh = f16 mirror of recon.
// Block = 8 waves -> 128x128 macro-tile. j-block DMA'd into LDS with async
// b128 ops (8/thread, zero VALU); A fragments are contiguous f16 global loads.
// ============================================================================
__global__ void pd_kernel(const _Float16* __restrict__ Xh, const float* __restrict__ sq,
                          float* __restrict__ PD)
{
    __shared__ _Float16 LX[128][136];        // 272B row stride: aligned + bank-spread
    const int tid  = threadIdx.x;
    const int lane = tid & 31;
    const int w    = tid >> 5;
    const int jg   = blockIdx.x;
    const int ig   = blockIdx.y;
    const int jbase = jg * 128;
    const int ti    = ig * 8 + w;
    const bool valid = (ti < TM);            // wave-uniform
    const int half = lane >> 4;
    const int l15  = lane & 15;

    // stage j-block: 128 rows x 256B = 2048 x 16B ops, 8 per thread
    for (int o = tid; o < 2048; o += 256) {
        const int jr = o >> 4, q = o & 15;
        const int j = jbase + jr;
        _Float16* lp = &LX[jr][q * 8];
        if (j < NN) {
            async_b128((unsigned)(size_t)lp,
                       (unsigned long long)(size_t)(Xh + (size_t)j * 128 + q * 8));
        } else {
#pragma unroll
            for (int i = 0; i < 8; ++i) lp[i] = (_Float16)0.f;  // zero-fill OOB rows
        }
    }
    wait_async0();
    __syncthreads();

    if (valid) {
        const int arow = ti * 16 + l15;
        v8f acc[8] = {};
#pragma unroll
        for (int k0 = 0; k0 < 128; k0 += 32) {
            v16h a;
            const _Float16* Ap = Xh + (size_t)arow * 128 + k0 + half * 8;
#pragma unroll
            for (int i = 0; i < 8; ++i) { a[i] = Ap[i]; a[i + 8] = Ap[i + 16]; }
#pragma unroll
            for (int jt = 0; jt < 8; ++jt) {
                v16h b;
                const _Float16* Bp = &LX[jt * 16 + l15][k0 + half * 8];
#pragma unroll
                for (int i = 0; i < 8; ++i) { b[i] = Bp[i]; b[i + 8] = Bp[i + 16]; }
                acc[jt] = __builtin_amdgcn_wmma_f32_16x16x32_f16(false, a, false, b,
                                                                 (short)0, acc[jt],
                                                                 false, false);
            }
        }
        const int rbase = ti * 16 + half * 8;
        float sr[8];
#pragma unroll
        for (int r = 0; r < 8; ++r) sr[r] = sq[rbase + r];
#pragma unroll
        for (int jt = 0; jt < 8; ++jt) {
            const int tj = jg * 8 + jt;
            if (tj >= TM) continue;          // wave-uniform
            const int col  = tj * 16 + l15;
            const float sc = sq[col];
#pragma unroll
            for (int r = 0; r < 8; ++r) {
                float d2 = sr[r] + sc - 2.f * acc[jt][r];
                PD[(size_t)(rbase + r) * NN + col] = sqrtf(fmaxf(d2, 1e-12f));
            }
        }
    }
}

// ---------- host-side launch ----------
extern "C" void kernel_launch(void* const* d_in, const int* in_sizes, int n_in,
                              void* d_out, int out_size, void* d_ws, size_t ws_size,
                              hipStream_t stream)
{
    const float* x   = (const float*)d_in[0];
    const int*   ei  = (const int*)  d_in[1];
    const float* W1  = (const float*)d_in[2];
    const float* a1s = (const float*)d_in[3];
    const float* a1d = (const float*)d_in[4];
    const float* b1  = (const float*)d_in[5];
    const float* W2  = (const float*)d_in[6];
    const float* a2s = (const float*)d_in[7];
    const float* a2d = (const float*)d_in[8];
    const float* b2  = (const float*)d_in[9];
    const float* Wmu = (const float*)d_in[10];
    const float* bmu = (const float*)d_in[11];
    const float* Wlv = (const float*)d_in[12];
    const float* blv = (const float*)d_in[13];
    const float* Wd1 = (const float*)d_in[14];
    const float* bd1 = (const float*)d_in[15];
    const float* Wd2 = (const float*)d_in[16];
    const float* bd2 = (const float*)d_in[17];
    const float* W3  = (const float*)d_in[18];
    const float* a3s = (const float*)d_in[19];
    const float* a3d = (const float*)d_in[20];
    const float* b3  = (const float*)d_in[21];

    float* out   = (float*)d_out;
    float* recon = out;                  // [N,128]
    float* mu    = out + 1280000;        // [N,64]
    float* lv    = out + 1920000;        // [N,64]
    float* pd    = out + 2560000;        // [N,N]

    // workspace carve (floats)
    float*    bufA  = (float*)d_ws;          // [N,1024] feature buffer
    float*    bufB  = bufA  + 10240000;      // [N,H*C] accumulator
    float*    bufC  = bufB  + 10240000;      // [N,256]
    float*    bufD  = bufC  + 2560000;       // [N,128]
    float*    exbuf = bufD  + 1280000;       // [ET*H]
    float*    al    = exbuf + 1320000;       // [N*H]
    float*    ar    = al    + 40000;
    unsigned* mx    = (unsigned*)(ar + 40000);
    float*    denom = (float*)mx + 40000;
    float*    zbuf  = denom + 40000;         // [N,64]
    float*    sqb   = zbuf  + 640000;        // [N]
    // f16 region (16B aligned: 26,450,000 floats from base)
    _Float16* Xh    = (_Float16*)(sqb + 10000);  // [N*128]
    _Float16* W1T   = Xh   + 1280000;        // [1024*128]
    _Float16* W2T   = W1T  + 131072;         // [512*256]
    _Float16* WmuT  = W2T  + 131072;         // [64*128]
    _Float16* WlvT  = WmuT + 8192;           // [64*128]
    _Float16* Wd1T  = WlvT + 8192;           // [128*64]
    _Float16* Wd2T  = Wd1T + 8192;           // [256*128]
    _Float16* W3T   = Wd2T + 32768;          // [512*256]

    auto convT = [&](const float* B, _Float16* BT, int K, int Nc) {
        convT_kernel<<<(K * Nc + 255) / 256, 256, 0, stream>>>(B, BT, K, Nc);
    };
    // pre-transpose+convert all GEMM weights to f16 (enables async DMA staging)
    convT(W1, W1T, 128, 1024);
    convT(W2, W2T, 256, 512);
    convT(Wmu, WmuT, 128, 64);
    convT(Wlv, WlvT, 128, 64);
    convT(Wd1, Wd1T, 64, 128);
    convT(Wd2, Wd2T, 128, 256);
    convT(W3, W3T, 256, 512);

    auto gemm = [&](const float* A, const _Float16* BT, const float* bias, float* C,
                    int M, int K, int Nc, int relu) {
        dim3 grid(Nc / 64, (M / 16 + 7) / 8);
        wmma_gemm_kernel<<<grid, 256, 0, stream>>>(A, BT, bias, C, M, K, Nc, relu);
    };

    auto gat = [&](const float* fin, int Fin, const _Float16* WmT, const float* as_,
                   const float* ad_, const float* bias, int C, float* outp, int relu) {
        const int HC = HH * C;
        const int NH = NN * HH;
        const int EH = ET * HH;
        gemm(fin, WmT, nullptr, bufA, NN, Fin, HC, 0);
        att_scores_kernel<<<(NH + 7) / 8, 256, 0, stream>>>(bufA, as_, ad_, al, ar, NH, C);
        fill_f32_kernel<<<(NN * HC + 255) / 256, 256, 0, stream>>>(bufB, 0.f, NN * HC);
        fill_f32_kernel<<<(NH + 255) / 256, 256, 0, stream>>>(denom, 0.f, NH);
        fill_u32_kernel<<<(NH + 255) / 256, 256, 0, stream>>>(mx, 0u, NH);
        edge_max_kernel<<<(EH + 255) / 256, 256, 0, stream>>>(ei, al, ar, mx);
        edge_exp_kernel<<<(EH + 255) / 256, 256, 0, stream>>>(ei, al, ar, mx, exbuf, denom);
        edge_scatter_kernel<<<(EH + 7) / 8, 256, 0, stream>>>(ei, exbuf, denom, bufA, bufB, C);
        head_mean_kernel<<<(NN * C + 255) / 256, 256, 0, stream>>>(bufB, bias, outp, C, relu);
    };

    // Encoder
    gat(x,    128, W1T, a1s, a1d, b1, 256, bufC, 1);  // conv1 -> hrelu1 [N,256]
    gat(bufC, 256, W2T, a2s, a2d, b2, 128, bufD, 1);  // conv2 -> h2     [N,128]
    gemm(bufD, WmuT, bmu, mu, NN, 128, 64, 0);
    gemm(bufD, WlvT, blv, lv, NN, 128, 64, 0);
    reparam_kernel<<<(640000 + 255) / 256, 256, 0, stream>>>(mu, lv, zbuf, 640000);
    // Decoder
    gemm(zbuf, Wd1T, bd1, bufD, NN, 64, 128, 1);      // d1 [N,128]
    gemm(bufD, Wd2T, bd2, bufC, NN, 128, 256, 1);     // d2 [N,256]
    gat(bufC, 256, W3T, a3s, a3d, b3, 128, recon, 0); // conv3 -> recon_x [N,128]
    // Pairwise distances
    tohalf_kernel<<<(1280000 + 255) / 256, 256, 0, stream>>>(recon, Xh, 1280000);
    sqnorm_kernel<<<(NN + 7) / 8, 256, 0, stream>>>(recon, sqb, 128);
    dim3 pg((TM + 7) / 8, (TM + 7) / 8);
    pd_kernel<<<pg, 256, 0, stream>>>(Xh, sqb, pd);
}
